// DCL_paper_7696581394729
// MI455X (gfx1250) — matrix operand order repaired and checked by reading, not compiled, not run on hardware
//
#include <hip/hip_runtime.h>
#include <hip/hip_bf16.h>
#include <math.h>

// ---------------------------------------------------------------------------
// DCL contrastive loss, fused: normalize -> f16 WMMA GEMM (u.vT and u.uT)
// with streaming fixed-max logsumexp epilogue -> deterministic reduction.
//
// B=8192, D=256. FLOPs = 68.7G (two 8192x8192x256 GEMMs), bytes ~16MB
// => compute bound; v_wmma_f32_16x16x32_f16 (8x FLOP/instr vs f32 WMMA).
// Each wave owns M=32 rows (two A strips in 128 VGPRs) so every B fragment
// pulled from LDS feeds TWO WMMAs -> halves LDS bytes/FLOP; 8 waves share
// each 16x256 B tile via double-buffered LDS (halves L2 traffic too).
// Fixed logsumexp max M0 = 1/T (cos sim <= 1) removes the online-max pass;
// pos_loss (uv diagonal) comes from a tiny dot kernel so the GEMM hot loop
// has no stores; diagonal masking only in the <=2 tiles where it can occur
// (wave-uniform scalar branches).
// ---------------------------------------------------------------------------

typedef __attribute__((ext_vector_type(16))) _Float16 v16h;
typedef __attribute__((ext_vector_type(8)))  _Float16 v8h;
typedef __attribute__((ext_vector_type(8)))  float    v8f;

#define NB        8192
#define DIM       256
#define INV_T     14.285714285714286f     // 1/0.07
#define LSE_M0    INV_T                   // fixed max: sim/T <= 1/T
#define LOG2E     1.4426950408889634f
#define LN2       0.6931471805599453f
#define C_SCALE   (INV_T * LOG2E)         // exp2-domain scale
#define C_BIAS    (-LSE_M0 * LOG2E)       // exp2-domain bias
#define CHUNKS    8                       // 8 column chunks over concat [uv|uu]
#define TILES_PER_CHUNK 128               // 2048 cols per chunk / 16
#define LDSPITCH  264                     // 256 + 8 halves: de-conflict banks

// --------------------------- row normalization -----------------------------
__global__ __launch_bounds__(256)
void normalize_rows(const float* __restrict__ src, _Float16* __restrict__ dst) {
    const int row = blockIdx.x;
    const int tid = threadIdx.x;              // 256 threads == DIM
    float x  = src[row * DIM + tid];
    float ss = x * x;
    ss += __shfl_xor(ss, 16, 32);
    ss += __shfl_xor(ss,  8, 32);
    ss += __shfl_xor(ss,  4, 32);
    ss += __shfl_xor(ss,  2, 32);
    ss += __shfl_xor(ss,  1, 32);
    __shared__ float red[8];
    if ((tid & 31) == 0) red[tid >> 5] = ss;
    __syncthreads();
    float tot = 0.f;
#pragma unroll
    for (int i = 0; i < 8; ++i) tot += red[i];
    const float inv = rsqrtf(tot);            // gaussian D=256 rows: |x|~16
    dst[row * DIM + tid] = (_Float16)(x * inv);
}

// ------------------- pos_loss: diag(sim_uv) = invT * <nu_i, nv_i> ----------
__global__ __launch_bounds__(256)
void pos_diag(const _Float16* __restrict__ nu, const _Float16* __restrict__ nv,
              float* __restrict__ pos) {
    const int lane = threadIdx.x & 31;
    const int row  = blockIdx.x * 8 + (threadIdx.x >> 5);
    v8h av = *(const v8h*)(nu + (size_t)row * DIM + lane * 8);
    v8h bv = *(const v8h*)(nv + (size_t)row * DIM + lane * 8);
    float acc = 0.f;
#pragma unroll
    for (int k = 0; k < 8; ++k) acc = fmaf((float)av[k], (float)bv[k], acc);
    acc += __shfl_xor(acc, 16, 32);
    acc += __shfl_xor(acc,  8, 32);
    acc += __shfl_xor(acc,  4, 32);
    acc += __shfl_xor(acc,  2, 32);
    acc += __shfl_xor(acc,  1, 32);
    if (lane == 0) pos[row] = acc * INV_T;
}

// ------------------ fused GEMM + streaming logsumexp -----------------------
// Grid: 256 WGs = 32 row-supertiles (256 rows) x 8 chunks. WG = 8 waves;
// wave w owns 32 rows (two 16-row strips); waves share each 16x256 B-tile
// via double-buffered LDS.
__global__ __launch_bounds__(256)
void dcl_gemm_lse(const _Float16* __restrict__ nu,
                  const _Float16* __restrict__ nv,
                  float* __restrict__ s_part) {  // [NB][CHUNKS]
    __shared__ _Float16 lds_b[2][16 * LDSPITCH];   // 2 x 8448 B

    const int tid      = threadIdx.x;
    const int lane     = tid & 31;
    const int wv       = tid >> 5;               // wave in WG (0..7)
    const int super    = blockIdx.x >> 3;        // row supertile (0..31)
    const int chunk    = blockIdx.x & 7;         // column chunk (0..7)
    const int row_base = super * 256 + wv * 32;  // 32 rows per wave
    const int half     = lane >> 4;              // hi/lo 16-lane group
    const int n        = lane & 15;              // output column within tile
    // wave-uniform copy for scalar branching on the (rare) diagonal tiles
    const int row_base_s = __builtin_amdgcn_readfirstlane(row_base);

    // --- A fragments: 2 strips x 16 rows x K=256 f16 in 128 VGPRs ----------
    // A layout (16x32 f16): lanes<16 hold K 0..15, lanes>=16 hold
    // K 16..31  => two contiguous b128 loads per k-chunk per lane.
    v16h a0[8], a1[8];
    {
        const _Float16* ar0 = nu + (size_t)(row_base + n) * DIM;
        const _Float16* ar1 = nu + (size_t)(row_base + 16 + n) * DIM;
#pragma unroll
        for (int kc = 0; kc < 8; ++kc) {
            const _Float16* ap0 = ar0 + kc * 32 + 16 * half;
            const _Float16* ap1 = ar1 + kc * 32 + 16 * half;
            v8h lo0 = *(const v8h*)ap0;
            v8h hi0 = *(const v8h*)(ap0 + 8);
            v8h lo1 = *(const v8h*)ap1;
            v8h hi1 = *(const v8h*)(ap1 + 8);
            a0[kc] = __builtin_shufflevector(lo0, hi0,
                        0,1,2,3,4,5,6,7,8,9,10,11,12,13,14,15);
            a1[kc] = __builtin_shufflevector(lo1, hi1,
                        0,1,2,3,4,5,6,7,8,9,10,11,12,13,14,15);
        }
    }

    // chunks 0..3 -> S_uv columns, 4..7 -> S_uu columns
    const bool is_uu = (chunk >= 4);
    const _Float16* bmat = is_uu ? nu : nv;
    const int col_chunk0 = (chunk & 3) * (TILES_PER_CHUNK * 16);

    float s0[8], s1[8];
#pragma unroll
    for (int r = 0; r < 8; ++r) { s0[r] = 0.f; s1[r] = 0.f; }

    // cooperative B-tile loader: thread t -> row t/16, 16-half slab t%16
    const int lr = tid >> 4;
    const int lp = tid & 15;
    const size_t ld_off = (size_t)lr * LDSPITCH + lp * 16;   // LDS halves
    const size_t g_off  = (size_t)lr * DIM + lp * 16;        // global halves

    // preload tile 0 into buffer 0
    {
        v16h t0 = *(const v16h*)(bmat + (size_t)col_chunk0 * DIM + g_off);
        *(v16h*)&lds_b[0][ld_off] = t0;
    }
    __syncthreads();

    for (int jt = 0; jt < TILES_PER_CHUNK; ++jt) {
        const int cur  = jt & 1;
        const int col0 = col_chunk0 + jt * 16;

        // issue next tile's global load early; hide latency behind WMMAs
        v16h nxt;
        const bool have_next = (jt + 1 < TILES_PER_CHUNK);
        if (have_next)
            nxt = *(const v16h*)(bmat + (size_t)(col0 + 16) * DIM + g_off);

        // --- 32x16 tile: 16 x v_wmma_f32_16x16x32_f16, each B fragment
        //     read once from LDS and fed to BOTH row strips ----------------
        // B k order used here, same as A (32x16 f16): lane n, lanes<16 K=0..15, lanes>=16
        // K=16..31 -> 16 contiguous halves per lane from the LDS row.
        const _Float16* brow = &lds_b[cur][(size_t)n * LDSPITCH + 16 * half];
        v8f c0 = {}, c1 = {};
#pragma unroll
        for (int kc = 0; kc < 8; ++kc) {
            v16h b = *(const v16h*)(brow + kc * 32);
            c0 = __builtin_amdgcn_wmma_f32_16x16x32_f16(
                     false, a0[kc], false, b, (short)0, c0, false, false);
            c1 = __builtin_amdgcn_wmma_f32_16x16x32_f16(
                     false, a1[kc], false, b, (short)0, c1, false, false);
        }

        // --- epilogue: fixed-max logsumexp in exp2 domain ------------------
        // C layout: VGPR r -> M = r + 8*half, N = lane%16. A 16-wide column
        // tile can only touch the diagonal of strip0 when col0 == row_base,
        // strip1 when col0 == row_base+16 (wave-uniform conditions).
        if (col0 == row_base_s) {
            const int diag_r = n - 8 * half;
#pragma unroll
            for (int r = 0; r < 8; ++r) {
                const float e = __builtin_amdgcn_exp2f(fmaf(c0[r], C_SCALE, C_BIAS));
                s0[r] += (r == diag_r) ? 0.f : e;
                s1[r] += __builtin_amdgcn_exp2f(fmaf(c1[r], C_SCALE, C_BIAS));
            }
        } else if (col0 == row_base_s + 16) {
            const int diag_r = n - 8 * half;
#pragma unroll
            for (int r = 0; r < 8; ++r) {
                s0[r] += __builtin_amdgcn_exp2f(fmaf(c0[r], C_SCALE, C_BIAS));
                const float e = __builtin_amdgcn_exp2f(fmaf(c1[r], C_SCALE, C_BIAS));
                s1[r] += (r == diag_r) ? 0.f : e;
            }
        } else {
#pragma unroll
            for (int r = 0; r < 8; ++r) {
                s0[r] += __builtin_amdgcn_exp2f(fmaf(c0[r], C_SCALE, C_BIAS));
                s1[r] += __builtin_amdgcn_exp2f(fmaf(c1[r], C_SCALE, C_BIAS));
            }
        }

        // stage next tile into the other buffer (its readers finished at the
        // barrier that ended iteration jt-1)
        if (have_next)
            *(v16h*)&lds_b[1 - cur][ld_off] = nxt;

        __syncthreads();
    }

    // reduce the 16 lanes that share each row, store per-(row,chunk) partial
#pragma unroll
    for (int r = 0; r < 8; ++r) {
        float t0 = s0[r], t1 = s1[r];
        t0 += __shfl_xor(t0, 1, 32);  t1 += __shfl_xor(t1, 1, 32);
        t0 += __shfl_xor(t0, 2, 32);  t1 += __shfl_xor(t1, 2, 32);
        t0 += __shfl_xor(t0, 4, 32);  t1 += __shfl_xor(t1, 4, 32);
        t0 += __shfl_xor(t0, 8, 32);  t1 += __shfl_xor(t1, 8, 32);
        if (n == 0) {
            const int grow = row_base + r + 8 * half;
            s_part[grow * CHUNKS + chunk]        = t0;
            s_part[(grow + 16) * CHUNKS + chunk] = t1;
        }
    }
}

// ------------------- deterministic final reduction -------------------------
__global__ __launch_bounds__(256)
void dcl_finalize(const float* __restrict__ s_part,
                  const float* __restrict__ pos,
                  float* __restrict__ out) {
    const int tid = threadIdx.x;
    float acc = 0.f;
    for (int row = tid; row < NB; row += 256) {
        float ssum = 0.f;
#pragma unroll
        for (int c = 0; c < CHUNKS; ++c) ssum += s_part[row * CHUNKS + c];
        // partials are 2^((x-M0)*log2e): lse = ln2*log2(sum) + M0
        const float neg = LN2 * __log2f(ssum) + LSE_M0;
        acc += neg - pos[row];
    }
    __shared__ float red[256];
    red[tid] = acc;
    __syncthreads();
    for (int off = 128; off >= 1; off >>= 1) {
        if (tid < off) red[tid] += red[tid + off];
        __syncthreads();
    }
    if (tid == 0) out[0] = red[0] / (float)NB;
}

// ---------------------------------------------------------------------------
extern "C" void kernel_launch(void* const* d_in, const int* in_sizes, int n_in,
                              void* d_out, int out_size, void* d_ws, size_t ws_size,
                              hipStream_t stream) {
    const float* u = (const float*)d_in[0];
    const float* v = (const float*)d_in[1];
    float* out = (float*)d_out;

    // workspace layout (~8.7 MB total)
    char* ws = (char*)d_ws;
    _Float16* nu     = (_Float16*)(ws);                            // 4 MB
    _Float16* nv     = (_Float16*)(ws + (size_t)4 * 1024 * 1024);  // 4 MB
    float*    s_part = (float*)   (ws + (size_t)8 * 1024 * 1024);  // 256 KB
    float*    pos    = (float*)   (ws + (size_t)8 * 1024 * 1024
                                      + (size_t)NB * CHUNKS * sizeof(float)); // 32 KB

    normalize_rows<<<NB, 256, 0, stream>>>(u, nu);
    normalize_rows<<<NB, 256, 0, stream>>>(v, nv);
    pos_diag<<<NB / 8, 256, 0, stream>>>(nu, nv, pos);
    dcl_gemm_lse<<<32 * CHUNKS, 256, 0, stream>>>(nu, nv, s_part);
    dcl_finalize<<<1, 256, 0, stream>>>(s_part, pos, out);
}
